// MambaModel_11759620457152
// MI455X (gfx1250) — matrix operand order repaired and checked by reading, compiled, not run on hardware
//
#include <hip/hip_runtime.h>
#include <hip/hip_bf16.h>

// ---------------------------------------------------------------------------
// Mamba forward on gfx1250. Reference reduces to ONE mixer layer (layer 3)
// applied to layer_norm(5*x) (residual only accumulates copies of x, LN is
// scale-invariant, y is overwritten each layer).
// GEMMs use bf16x3 split precision on V_WMMA_F32_16X16X32_BF16 (XDL pipe):
//   A*B ~= Ah*Bh + Ah*Bl + Al*Bh   (f32 accumulate, ~fp32 accuracy)
// Weights are [N][K] row-major already -> no transposes; pre-split to bf16
// hi/lo once. Activations split during LDS staging.
// ---------------------------------------------------------------------------

typedef float v8f   __attribute__((ext_vector_type(8)));
typedef __bf16 v2bf  __attribute__((ext_vector_type(2)));
typedef __bf16 v8bf  __attribute__((ext_vector_type(8)));
typedef __bf16 v16bf __attribute__((ext_vector_type(16)));

#define DIMM   1024
#define DINNER 2048
#define LSEQ   2048
#define NBATCH 2
#define NROWS  (NBATCH * LSEQ)   // 4096 = M for all GEMMs
#define DSTATE 16
#define DTRANK 64
#define XDBLC  96                // DT_RANK + 2*D_STATE

// ---------------------------- weight split to bf16 hi/lo --------------------
__global__ __launch_bounds__(256)
void split_bf16_kernel(const float* __restrict__ in, __bf16* __restrict__ hi,
                       __bf16* __restrict__ lo, int n) {
  int i = blockIdx.x * 256 + threadIdx.x;
  if (i < n) {
    float v = in[i];
    __bf16 h = (__bf16)v;
    hi[i] = h;
    lo[i] = (__bf16)(v - (float)h);
  }
}

// ---------------------------- layernorm of 5*x ------------------------------
__global__ __launch_bounds__(256)
void ln_kernel(const float* __restrict__ x, const float* __restrict__ w,
               const float* __restrict__ b, float* __restrict__ h) {
  __shared__ float red[256];
  const int row = blockIdx.x;
  const int tid = threadIdx.x;
  const float* xr = x + (size_t)row * DIMM;
  float v[4];
  float s = 0.f;
  #pragma unroll
  for (int i = 0; i < 4; ++i) { v[i] = 5.0f * xr[tid + i * 256]; s += v[i]; }
  red[tid] = s; __syncthreads();
  for (int o = 128; o > 0; o >>= 1) { if (tid < o) red[tid] += red[tid + o]; __syncthreads(); }
  const float mu = red[0] * (1.f / DIMM);
  __syncthreads();
  float vs = 0.f;
  #pragma unroll
  for (int i = 0; i < 4; ++i) { float dl = v[i] - mu; vs += dl * dl; }
  red[tid] = vs; __syncthreads();
  for (int o = 128; o > 0; o >>= 1) { if (tid < o) red[tid] += red[tid + o]; __syncthreads(); }
  const float inv = rsqrtf(red[0] * (1.f / DIMM) + 1e-5f);
  float* hr = h + (size_t)row * DIMM;
  #pragma unroll
  for (int i = 0; i < 4; ++i) {
    int c = tid + i * 256;
    hr[c] = (v[i] - mu) * inv * w[c] + b[c];
  }
}

// ---------------------------- bf16x3 WMMA GEMM ------------------------------
// C[M,N] = A[M,K](f32, split on the fly) * W[N,K](pre-split bf16 hi/lo)
// EPI: 0 plain store, 1 softplus(x + bias[n]), 2 split store n<2048->C0 else C1
#define BM  64
#define BN  64
#define BK  32
#define LDT 40   // padded LDS row (bf16): 80B stride -> conflict-free b128 reads

__device__ __forceinline__ v16bf ld16(const __bf16* p0, const __bf16* p1) {
  v8bf a = *(const v8bf*)p0;
  v8bf b = *(const v8bf*)p1;
  return __builtin_shufflevector(a, b, 0, 1, 2, 3, 4, 5, 6, 7,
                                 8, 9, 10, 11, 12, 13, 14, 15);
}

template <int EPI>
__global__ __launch_bounds__(256)
void gemm_bf16x3(const float* __restrict__ A, int lda,
                 const __bf16* __restrict__ Bhi, const __bf16* __restrict__ Blo,
                 int ldb, float* __restrict__ C0, float* __restrict__ C1, int ldc,
                 const float* __restrict__ bias, int M, int N, int K) {
  __shared__ __bf16 Ah[BM][LDT];
  __shared__ __bf16 Al[BM][LDT];
  __shared__ __bf16 Bh[BN][LDT];
  __shared__ __bf16 Bl[BN][LDT];
  const int tid  = threadIdx.x;
  const int bm   = blockIdx.y * BM;
  const int bn   = blockIdx.x * BN;
  const int wave = tid >> 5;
  const int lane = tid & 31;
  const int lrow = lane & 15;
  const int lhi  = lane >> 4;
  const int wm   = (wave & 3) << 4;   // wave M offset
  const int wn   = (wave >> 2) << 5;  // wave N offset (0 or 32)

  v8f acc0 = {0.f, 0.f, 0.f, 0.f, 0.f, 0.f, 0.f, 0.f};
  v8f acc1 = {0.f, 0.f, 0.f, 0.f, 0.f, 0.f, 0.f, 0.f};

  for (int k0 = 0; k0 < K; k0 += BK) {
    if (k0 + BK < K) {  // prefetch next K-chunk (global_prefetch_b8)
      __builtin_prefetch(&A[(size_t)(bm + (tid >> 5)) * lda + k0 + BK + (tid & 31)], 0, 3);
      __builtin_prefetch(&Bhi[(size_t)(bn + (tid >> 5)) * ldb + k0 + BK + (tid & 31)], 0, 3);
    }
    // stage tiles: 1024 bf16-pairs per matrix, 4 dword-packed writes/thread
    #pragma unroll
    for (int i = 0; i < 4; ++i) {
      const int p  = tid + i * 256;   // pair index over 64x16
      const int r  = p >> 4;          // tile row
      const int kp = (p & 15) * 2;    // even k
      const size_t ga = (size_t)(bm + r) * lda + (k0 + kp);
      float va0 = A[ga], va1 = A[ga + 1];
      __bf16 h0 = (__bf16)va0, h1 = (__bf16)va1;
      v2bf ph = {h0, h1};
      v2bf pl = {(__bf16)(va0 - (float)h0), (__bf16)(va1 - (float)h1)};
      *(v2bf*)&Ah[r][kp] = ph;
      *(v2bf*)&Al[r][kp] = pl;
      const int n = bn + r;
      v2bf wh = {(__bf16)0.f, (__bf16)0.f}, wl = wh;
      if (n < N) {
        wh = *(const v2bf*)&Bhi[(size_t)n * ldb + (k0 + kp)];
        wl = *(const v2bf*)&Blo[(size_t)n * ldb + (k0 + kp)];
      }
      *(v2bf*)&Bh[r][kp] = wh;
      *(v2bf*)&Bl[r][kp] = wl;
    }
    __syncthreads();
    // fragments use contiguous k order (same for A and B):
    //  A: elems 0-15 <- K=16*g..16*g+15  (g = lane>>4)
    //  B: elems 0-15 <- K=16*g..16*g+15
    const int ma = wm + lrow;
    v16bf ah = ld16(&Ah[ma][16 * lhi], &Ah[ma][16 * lhi + 8]);
    v16bf al = ld16(&Al[ma][16 * lhi], &Al[ma][16 * lhi + 8]);
    const int n0 = wn + lrow, n1 = wn + 16 + lrow;
    v16bf b0h = ld16(&Bh[n0][16 * lhi], &Bh[n0][16 * lhi + 8]);
    v16bf b0l = ld16(&Bl[n0][16 * lhi], &Bl[n0][16 * lhi + 8]);
    v16bf b1h = ld16(&Bh[n1][16 * lhi], &Bh[n1][16 * lhi + 8]);
    v16bf b1l = ld16(&Bl[n1][16 * lhi], &Bl[n1][16 * lhi + 8]);
    acc0 = __builtin_amdgcn_wmma_f32_16x16x32_bf16(false, ah, false, b0h, (short)0, acc0, false, false);
    acc0 = __builtin_amdgcn_wmma_f32_16x16x32_bf16(false, ah, false, b0l, (short)0, acc0, false, false);
    acc0 = __builtin_amdgcn_wmma_f32_16x16x32_bf16(false, al, false, b0h, (short)0, acc0, false, false);
    acc1 = __builtin_amdgcn_wmma_f32_16x16x32_bf16(false, ah, false, b1h, (short)0, acc1, false, false);
    acc1 = __builtin_amdgcn_wmma_f32_16x16x32_bf16(false, ah, false, b1l, (short)0, acc1, false, false);
    acc1 = __builtin_amdgcn_wmma_f32_16x16x32_bf16(false, al, false, b1h, (short)0, acc1, false, false);
    __syncthreads();
  }

  #pragma unroll
  for (int r = 0; r < 8; ++r) {
    const int m = bm + wm + r + 8 * lhi;
    #pragma unroll
    for (int t = 0; t < 2; ++t) {
      const int n = bn + wn + t * 16 + lrow;
      const float v = (t == 0) ? acc0[r] : acc1[r];
      if (EPI == 0) {
        if (n < N) C0[(size_t)m * ldc + n] = v;
      } else if (EPI == 1) {
        if (n < N) {
          float u = v + bias[n];
          C0[(size_t)m * ldc + n] = (u > 20.f) ? u : log1pf(__expf(u));
        }
      } else {  // split: n<DINNER -> C0 (xm), else C1 (z)
        if (n < DINNER) C0[(size_t)m * ldc + n] = v;
        else            C1[(size_t)m * ldc + (n - DINNER)] = v;
      }
    }
  }
}

// ---------------------------- depthwise causal conv + silu ------------------
__global__ __launch_bounds__(256)
void conv_silu_kernel(const float* __restrict__ xm, const float* __restrict__ cw,
                      const float* __restrict__ cb, float* __restrict__ xc) {
  const int idx = blockIdx.x * 256 + threadIdx.x;  // over NROWS*DINNER
  const int d  = idx & (DINNER - 1);
  const int bl = idx >> 11;
  const int l  = bl & (LSEQ - 1);
  const size_t base = (size_t)bl * DINNER + d;
  float acc = cb[d];
  const float w0 = cw[d * 4 + 0], w1 = cw[d * 4 + 1];
  const float w2 = cw[d * 4 + 2], w3 = cw[d * 4 + 3];
  if (l >= 3) acc += w0 * xm[base - 3 * (size_t)DINNER];
  if (l >= 2) acc += w1 * xm[base - 2 * (size_t)DINNER];
  if (l >= 1) acc += w2 * xm[base - 1 * (size_t)DINNER];
  acc += w3 * xm[base];
  xc[base] = acc / (1.f + __expf(-acc));  // silu
}

// ---------------------------- selective scan --------------------------------
__global__ __launch_bounds__(256)
void scan_kernel(const float* __restrict__ dtsp, const float* __restrict__ xc,
                 const float* __restrict__ xdbl, const float* __restrict__ zb,
                 const float* __restrict__ a_log, const float* __restrict__ dpar,
                 float* __restrict__ yb) {
  __shared__ float bc[64][32];  // [timestep][B(0..15) | C(16..31)]
  const int tid = threadIdx.x;
  const int b   = blockIdx.x >> 3;
  const int d   = (blockIdx.x & 7) * 256 + tid;

  float Arow[DSTATE], hst[DSTATE];
  #pragma unroll
  for (int s = 0; s < DSTATE; ++s) {
    Arow[s] = -__expf(a_log[(size_t)d * DSTATE + s]);
    hst[s] = 0.f;
  }
  const float Dd = dpar[d];
  const size_t rowbase = (size_t)b * LSEQ * DINNER;

  for (int t0 = 0; t0 < LSEQ; t0 += 64) {
    __syncthreads();
    #pragma unroll
    for (int i = 0; i < 8; ++i) {
      int e = tid + i * 256;
      int tt = e >> 5, cc = e & 31;
      bc[tt][cc] = xdbl[(size_t)(b * LSEQ + t0 + tt) * XDBLC + DTRANK + cc];
    }
    __syncthreads();
    for (int tt = 0; tt < 64; ++tt) {
      const size_t off = rowbase + (size_t)(t0 + tt) * DINNER + d;
      const float dtv = dtsp[off];
      const float xv  = xc[off];
      const float zv  = zb[off];
      const float dtx = dtv * xv;
      float y = 0.f;
      #pragma unroll
      for (int s = 0; s < DSTATE; ++s) {
        float dA = __expf(dtv * Arow[s]);
        float hn = dA * hst[s] + dtx * bc[tt][s];
        hst[s] = hn;
        y += hn * bc[tt][16 + s];
      }
      yb[off] = (y + Dd * xv) * (zv / (1.f + __expf(-zv)));
    }
  }
}

// ---------------------------------------------------------------------------
extern "C" void kernel_launch(void* const* d_in, const int* in_sizes, int n_in,
                              void* d_out, int out_size, void* d_ws, size_t ws_size,
                              hipStream_t stream) {
  const int LAST = 3;
  const float* x    = (const float*)d_in[0];
  const float* lnw  = (const float*)d_in[1]  + LAST * DIMM;
  const float* lnb  = (const float*)d_in[2]  + LAST * DIMM;
  const float* inw  = (const float*)d_in[3]  + (size_t)LAST * 2 * DINNER * DIMM;
  const float* cw   = (const float*)d_in[4]  + (size_t)LAST * DINNER * 4;
  const float* cb   = (const float*)d_in[5]  + LAST * DINNER;
  const float* xpw  = (const float*)d_in[6]  + (size_t)LAST * XDBLC * DINNER;
  const float* dtw  = (const float*)d_in[7]  + (size_t)LAST * DINNER * DTRANK;
  const float* dtb  = (const float*)d_in[8]  + LAST * DINNER;
  const float* alog = (const float*)d_in[9]  + (size_t)LAST * DINNER * DSTATE;
  const float* dpar = (const float*)d_in[10] + LAST * DINNER;
  const float* outw = (const float*)d_in[11] + (size_t)LAST * DIMM * DINNER;
  float* out = (float*)d_out;

  const size_t nInW  = (size_t)2 * DINNER * DIMM;  // 4194304
  const size_t nXpW  = (size_t)XDBLC * DINNER;     // 196608
  const size_t nDtW  = (size_t)DINNER * DTRANK;    // 131072
  const size_t nOutW = (size_t)DIMM * DINNER;      // 2097152

  __bf16* bp = (__bf16*)d_ws;
  __bf16* w_in_hi  = bp; bp += nInW;
  __bf16* w_in_lo  = bp; bp += nInW;
  __bf16* w_xp_hi  = bp; bp += nXpW;
  __bf16* w_xp_lo  = bp; bp += nXpW;
  __bf16* w_dt_hi  = bp; bp += nDtW;
  __bf16* w_dt_lo  = bp; bp += nDtW;
  __bf16* w_out_hi = bp; bp += nOutW;
  __bf16* w_out_lo = bp; bp += nOutW;
  float* fp = (float*)bp;  // even element count -> 4B aligned
  float* hbuf  = fp; fp += (size_t)NROWS * DIMM;    // [4096][1024]
  float* xmbuf = fp; fp += (size_t)NROWS * DINNER;  // [4096][2048] (reused for dt)
  float* zbuf  = fp; fp += (size_t)NROWS * DINNER;  // [4096][2048] (reused for y)
  float* xcbuf = fp; fp += (size_t)NROWS * DINNER;  // [4096][2048]
  float* xdbl  = fp; fp += (size_t)NROWS * XDBLC;   // [4096][96]
  const size_t need = (size_t)((char*)fp - (char*)d_ws);
  if (ws_size < need) return;

  // 1) split weights into bf16 hi/lo (layouts are already [N][K])
  split_bf16_kernel<<<(int)((nInW  + 255) / 256), 256, 0, stream>>>(inw,  w_in_hi,  w_in_lo,  (int)nInW);
  split_bf16_kernel<<<(int)((nXpW  + 255) / 256), 256, 0, stream>>>(xpw,  w_xp_hi,  w_xp_lo,  (int)nXpW);
  split_bf16_kernel<<<(int)((nDtW  + 255) / 256), 256, 0, stream>>>(dtw,  w_dt_hi,  w_dt_lo,  (int)nDtW);
  split_bf16_kernel<<<(int)((nOutW + 255) / 256), 256, 0, stream>>>(outw, w_out_hi, w_out_lo, (int)nOutW);

  // 2) h = layernorm(5*x)
  ln_kernel<<<NROWS, 256, 0, stream>>>(x, lnw, lnb, hbuf);

  // 3) xz = h @ in_w^T  (split into xm / z)
  gemm_bf16x3<2><<<dim3((2 * DINNER) / BN, NROWS / BM), 256, 0, stream>>>(
      hbuf, DIMM, w_in_hi, w_in_lo, DIMM, xmbuf, zbuf, DINNER, nullptr,
      NROWS, 2 * DINNER, DIMM);

  // 4) depthwise causal conv + silu
  conv_silu_kernel<<<(NROWS * DINNER) / 256, 256, 0, stream>>>(xmbuf, cw, cb, xcbuf);

  // 5) x_dbl = xc @ x_proj^T  (N=96, bounds-guarded)
  gemm_bf16x3<0><<<dim3((XDBLC + BN - 1) / BN, NROWS / BM), 256, 0, stream>>>(
      xcbuf, DINNER, w_xp_hi, w_xp_lo, DINNER, xdbl, nullptr, XDBLC, nullptr,
      NROWS, XDBLC, DINNER);

  // 6) dt = softplus(x_dbl[:, :64] @ dt_w^T + dt_b)   (overwrites xm buffer)
  gemm_bf16x3<1><<<dim3(DINNER / BN, NROWS / BM), 256, 0, stream>>>(
      xdbl, XDBLC, w_dt_hi, w_dt_lo, DTRANK, xmbuf, nullptr, DINNER, dtb,
      NROWS, DINNER, DTRANK);

  // 7) selective scan; y written in-place over z buffer
  scan_kernel<<<NBATCH * (DINNER / 256), 256, 0, stream>>>(
      xmbuf, xcbuf, xdbl, zbuf, alog, dpar, zbuf);

  // 8) out = y @ out_w^T
  gemm_bf16x3<0><<<dim3(DIMM / BN, NROWS / BM), 256, 0, stream>>>(
      zbuf, DINNER, w_out_hi, w_out_lo, DINNER, out, nullptr, DIMM, nullptr,
      NROWS, DIMM, DINNER);
}
